// QuantizationLayer_25589415149761
// MI455X (gfx1250) — matrix operand order repaired and checked by reading, not compiled, not run on hardware
//
#include <hip/hip_runtime.h>

#define EMBED_D   64
#define NUM_CODES 1024
#define ROW_US    72          // padded LDS row stride: 64 data + 8 pad ushorts (144 B)
#define ROW_B     144

typedef __attribute__((ext_vector_type(16))) __bf16          v16bf;
typedef __attribute__((ext_vector_type(16))) unsigned short  v16us;
typedef __attribute__((ext_vector_type(8)))  float           v8f;
typedef __attribute__((ext_vector_type(4)))  unsigned int    u32x4;
typedef __attribute__((ext_vector_type(8)))  int             i32x8;
typedef __attribute__((ext_vector_type(4)))  int             i32x4;

#if __has_builtin(__builtin_amdgcn_tensor_load_to_lds) && \
    __has_builtin(__builtin_amdgcn_s_wait_tensorcnt)
#define USE_TDM 1
#endif

static __device__ __forceinline__ unsigned short f32_bf16_rn(float f) {
    unsigned u = __float_as_uint(f);
    unsigned r = 0x7FFFu + ((u >> 16) & 1u);   // round-to-nearest-even
    return (unsigned short)((u + r) >> 16);
}
static __device__ __forceinline__ float bf16_f32(unsigned short h) {
    return __uint_as_float(((unsigned)h) << 16);
}

// ---------------------------------------------------------------------------
// Kernel 0: split codebook into bf16 hi/lo planes + per-code squared norms.
// ---------------------------------------------------------------------------
__global__ __launch_bounds__(32) void vq_prep_kernel(
    const float* __restrict__ emb,
    unsigned short* __restrict__ ehi,
    unsigned short* __restrict__ elo,
    float* __restrict__ enorm)
{
    const int k    = blockIdx.x;
    const int lane = threadIdx.x & 31;
    float s = 0.0f;
    #pragma unroll
    for (int i = 0; i < 2; ++i) {
        int d = lane + 32 * i;
        float f = emb[(size_t)k * EMBED_D + d];
        unsigned short hb = f32_bf16_rn(f);
        unsigned short lb = f32_bf16_rn(f - bf16_f32(hb));
        ehi[(size_t)k * EMBED_D + d] = hb;
        elo[(size_t)k * EMBED_D + d] = lb;
        s += f * f;
    }
    #pragma unroll
    for (int m = 16; m >= 1; m >>= 1) s += __shfl_xor(s, m, 32);
    if (lane == 0) enorm[k] = s;
}

#ifdef USE_TDM
static __device__ __forceinline__ unsigned lds_offset_of(const void* p) {
    return (unsigned)(unsigned long long)
        (__attribute__((address_space(3))) const void*)p;
}
// One TDM descriptor moves a full 1024x128B plane into LDS, padding each
// 32-dword row with 4 dwords so LDS rows land on a 144 B (bank-conflict-free)
// stride.  data_size = 4 B.
static __device__ __forceinline__ void tdm_load_plane(const void* gsrc,
                                                      unsigned lds_off) {
    unsigned long long ga = (unsigned long long)gsrc;
    u32x4 g0;
    g0[0] = 1u;                                   // count=1, user descriptor
    g0[1] = lds_off;                              // lds_addr
    g0[2] = (unsigned)(ga & 0xFFFFFFFFull);       // global_addr[31:0]
    g0[3] = (unsigned)((ga >> 32) & 0x01FFFFFFull) | (2u << 30);  // [56:32]|type=2
    i32x8 g1;
    g1[0] = (int)((2u << 16)      // data_size = 4B
                | (1u << 20)      // pad_enable
                | (4u << 22)      // pad_interval: 32 dwords (one row)
                | (3u << 25));    // pad_amount: 4 dwords (16 B)
    g1[1] = (int)(32u << 16);     // tensor_dim0 = 32 dwords
    g1[2] = (int)(1024u << 16);   // tensor_dim1 = 1024 rows
    g1[3] = (int)(32u << 16);     // tile_dim0 = 32 dwords
    g1[4] = (int)1024;            // tile_dim1 = 1024 rows, tile_dim2 = 0
    g1[5] = 32;                   // tensor_dim0_stride = 32 dwords
    g1[6] = 0;
    g1[7] = 0;
    i32x4 z4 = {0, 0, 0, 0};
#if __clang_major__ >= 23
    i32x8 z8 = {0, 0, 0, 0, 0, 0, 0, 0};
    __builtin_amdgcn_tensor_load_to_lds(g0, g1, z4, z4, z8, 0);
#else
    __builtin_amdgcn_tensor_load_to_lds(g0, g1, z4, z4, 0);
#endif
}
#endif

// load one 32-byte B fragment (16 bf16) from LDS via two 16 B ds reads
static __device__ __forceinline__ v16bf ld_bfrag(const unsigned char* p) {
    struct S { uint4 a, b; } s;
    s.a = *(const uint4*)(p);
    s.b = *(const uint4*)(p + 16);
    return __builtin_bit_cast(v16bf, s);
}

// ---------------------------------------------------------------------------
// Kernel 1: TDM-staged, LDS-resident codebook; fused WMMA scoring + argmin +
// gather.  Each wave owns TWO 16-token tiles (32 tokens), reusing every B
// tile twice -> 12 v_wmma per code tile.  512 threads = 16 waves / block.
// ---------------------------------------------------------------------------
__global__ __launch_bounds__(512) void vq_wmma_kernel(
    const float* __restrict__ x,
    const float* __restrict__ emb,
    const unsigned short* __restrict__ ehi,
    const unsigned short* __restrict__ elo,
    const float* __restrict__ enorm,
    float* __restrict__ outq,
    float* __restrict__ outst)
{
    __shared__ unsigned short lds_ehi[NUM_CODES * ROW_US];   // 144 KB
    __shared__ unsigned short lds_elo[NUM_CODES * ROW_US];   // 144 KB
    __shared__ float          lds_en[NUM_CODES];             //   4 KB

    const int tid  = threadIdx.x;
    const int lane = tid & 31;
    const int wave = tid >> 5;
    const int h    = lane >> 4;     // lane-half selector (ISA layout)
    const int c    = lane & 15;     // column / row-within-group

    // ---- stage codebook planes into LDS ------------------------------------
#ifdef USE_TDM
    if (wave == 0) {
        tdm_load_plane(ehi, lds_offset_of(lds_ehi));
        tdm_load_plane(elo, lds_offset_of(lds_elo));
        __builtin_amdgcn_s_wait_tensorcnt(0);
    }
#else
    for (int i = tid; i < NUM_CODES * 8; i += 512) {   // 16 B chunks
        int row = i >> 3, piece = i & 7;
        uint4 a = *(const uint4*)(ehi + (size_t)row * EMBED_D + piece * 8);
        uint4 b = *(const uint4*)(elo + (size_t)row * EMBED_D + piece * 8);
        *(uint4*)((unsigned char*)lds_ehi + row * ROW_B + piece * 16) = a;
        *(uint4*)((unsigned char*)lds_elo + row * ROW_B + piece * 16) = b;
    }
#endif
    for (int i = tid; i < NUM_CODES; i += 512) lds_en[i] = enorm[i];
    __syncthreads();

    // ---- load + split the two A tiles (2 x 16 tokens x 64 dims) ------------
    // 16-bit A 16x32 layout: elem j -> K = 16h+j (contiguous, matches B)
    const int tbase = blockIdx.x * 512 + wave * 32;
    v16us u[8];   // [tile][dhalf][hi/lo] packed: 0..3 tile0, 4..7 tile1
    #pragma unroll
    for (int tile = 0; tile < 2; ++tile) {
        const float* xrow = x + (size_t)(tbase + tile * 16 + c) * EMBED_D;
        #pragma unroll
        for (int j = 0; j < 16; ++j) {
            int d = 16 * h + j;
            float f0 = xrow[d];
            float f1 = xrow[d + 32];
            unsigned short hb0 = f32_bf16_rn(f0);
            unsigned short hb1 = f32_bf16_rn(f1);
            u[tile * 4 + 0][j] = hb0;
            u[tile * 4 + 1][j] = f32_bf16_rn(f0 - bf16_f32(hb0));
            u[tile * 4 + 2][j] = hb1;
            u[tile * 4 + 3][j] = f32_bf16_rn(f1 - bf16_f32(hb1));
        }
    }
    const v16bf A0h = __builtin_bit_cast(v16bf, u[0]);
    const v16bf A0l = __builtin_bit_cast(v16bf, u[1]);
    const v16bf A1h = __builtin_bit_cast(v16bf, u[2]);
    const v16bf A1l = __builtin_bit_cast(v16bf, u[3]);
    const v16bf C0h = __builtin_bit_cast(v16bf, u[4]);
    const v16bf C0l = __builtin_bit_cast(v16bf, u[5]);
    const v16bf C1h = __builtin_bit_cast(v16bf, u[6]);
    const v16bf C1l = __builtin_bit_cast(v16bf, u[7]);

    float bestv[16];
    int   bidx[16];
    #pragma unroll
    for (int v = 0; v < 16; ++v) { bestv[v] = 3.4e38f; bidx[v] = 0; }

    for (int t = 0; t < 64; ++t) {
        const int kbase = t * 16;
        // B 32x16 layout: lane column = c, elem j -> K = 16h + j (contiguous).
        const unsigned char* ph =
            (const unsigned char*)lds_ehi + (kbase + c) * ROW_B + 32 * h;
        const unsigned char* pl =
            (const unsigned char*)lds_elo + (kbase + c) * ROW_B + 32 * h;
        const v16bf B0h = ld_bfrag(ph);
        const v16bf B0l = ld_bfrag(pl);
        const v16bf B1h = ld_bfrag(ph + 64);
        const v16bf B1l = ld_bfrag(pl + 64);

        // bf16x3: x.e ~= xh.eh + xh.el + xl.eh  (four independent chains)
        v8f p0 = {}, p1 = {}, q0 = {}, q1 = {};
        p0 = __builtin_amdgcn_wmma_f32_16x16x32_bf16(false, A0h, false, B0h, (short)0, p0, false, false);
        p1 = __builtin_amdgcn_wmma_f32_16x16x32_bf16(false, A1h, false, B1h, (short)0, p1, false, false);
        q0 = __builtin_amdgcn_wmma_f32_16x16x32_bf16(false, C0h, false, B0h, (short)0, q0, false, false);
        q1 = __builtin_amdgcn_wmma_f32_16x16x32_bf16(false, C1h, false, B1h, (short)0, q1, false, false);
        p0 = __builtin_amdgcn_wmma_f32_16x16x32_bf16(false, A0h, false, B0l, (short)0, p0, false, false);
        p1 = __builtin_amdgcn_wmma_f32_16x16x32_bf16(false, A1h, false, B1l, (short)0, p1, false, false);
        q0 = __builtin_amdgcn_wmma_f32_16x16x32_bf16(false, C0h, false, B0l, (short)0, q0, false, false);
        q1 = __builtin_amdgcn_wmma_f32_16x16x32_bf16(false, C1h, false, B1l, (short)0, q1, false, false);
        p0 = __builtin_amdgcn_wmma_f32_16x16x32_bf16(false, A0l, false, B0h, (short)0, p0, false, false);
        p1 = __builtin_amdgcn_wmma_f32_16x16x32_bf16(false, A1l, false, B1h, (short)0, p1, false, false);
        q0 = __builtin_amdgcn_wmma_f32_16x16x32_bf16(false, C0l, false, B0h, (short)0, q0, false, false);
        q1 = __builtin_amdgcn_wmma_f32_16x16x32_bf16(false, C1l, false, B1h, (short)0, q1, false, false);

        const float en   = lds_en[kbase + c];
        const int   kcol = kbase + c;
        #pragma unroll
        for (int v = 0; v < 8; ++v) {
            float s0 = en - 2.0f * (p0[v] + p1[v]);
            float s1 = en - 2.0f * (q0[v] + q1[v]);
            bool b0 = s0 < bestv[v];             // strict < keeps earliest index
            bool b1 = s1 < bestv[v + 8];
            bestv[v]     = b0 ? s0   : bestv[v];
            bidx[v]      = b0 ? kcol : bidx[v];
            bestv[v + 8] = b1 ? s1   : bestv[v + 8];
            bidx[v + 8]  = b1 ? kcol : bidx[v + 8];
        }
    }

    // ---- argmin across the 16-lane column group (both halves independent) --
    #pragma unroll
    for (int m = 1; m < 16; m <<= 1) {
        #pragma unroll
        for (int v = 0; v < 16; ++v) {
            float os = __shfl_xor(bestv[v], m, 32);
            int   oi = __shfl_xor(bidx[v],  m, 32);
            bool take = (os < bestv[v]) || (os == bestv[v] && oi < bidx[v]);
            bestv[v] = take ? os : bestv[v];
            bidx[v]  = take ? oi : bidx[v];
        }
    }

    // ---- gather winning rows; 16 lanes x float4 cover one 64-float row -----
    #pragma unroll
    for (int v = 0; v < 16; ++v) {
        const int tile  = v >> 3;
        const int token = tbase + tile * 16 + 8 * h + (v & 7);
        const int k     = bidx[v];
        const float4 e4 = *(const float4*)(emb + (size_t)k * EMBED_D + 4 * c);
        const float4 x4 = *(const float4*)(x + (size_t)token * EMBED_D + 4 * c);
        float4 st;
        st.x = (e4.x - x4.x) + x4.x;
        st.y = (e4.y - x4.y) + x4.y;
        st.z = (e4.z - x4.z) + x4.z;
        st.w = (e4.w - x4.w) + x4.w;
        *(float4*)(outq  + (size_t)token * EMBED_D + 4 * c) = e4;
        *(float4*)(outst + (size_t)token * EMBED_D + 4 * c) = st;
    }
}

extern "C" void kernel_launch(void* const* d_in, const int* in_sizes, int n_in,
                              void* d_out, int out_size, void* d_ws, size_t ws_size,
                              hipStream_t stream) {
    const float* x   = (const float*)d_in[0];
    const float* emb = (const float*)d_in[1];
    const int n_tokens = in_sizes[0] / EMBED_D;          // 262144

    // workspace: bf16-hi plane | bf16-lo plane | norms  (~260 KB, L2-resident)
    unsigned short* ehi   = (unsigned short*)d_ws;
    unsigned short* elo   = ehi + (size_t)NUM_CODES * EMBED_D;
    float*          enorm = (float*)(elo + (size_t)NUM_CODES * EMBED_D);

    float* outq  = (float*)d_out;
    float* outst = outq + (size_t)n_tokens * EMBED_D;

    vq_prep_kernel<<<NUM_CODES, 32, 0, stream>>>(emb, ehi, elo, enorm);
    vq_wmma_kernel<<<n_tokens / 512, 512, 0, stream>>>(x, emb, ehi, elo, enorm,
                                                       outq, outst);
}
